// FeedbackAlignmentEqProp_46084999086182
// MI455X (gfx1250) — matrix operand order repaired and checked by reading, not compiled, not run on hardware
//
#include <hip/hip_runtime.h>
#include <hip/hip_bf16.h>
#include <math.h>

typedef __attribute__((ext_vector_type(16))) __bf16 v16bf;
typedef __attribute__((ext_vector_type(8)))  __bf16 v8bf;
typedef __attribute__((ext_vector_type(8)))  float  v8f;

#define ALPHA_C 0.5f
#define EPS_C   1e-12f

#define EPI_XPROJ 0
#define EPI_STEP  1
#define EPI_HEAD  2

// ---------------------------------------------------------------------------
// CDNA5 async global->LDS copy (ASYNCcnt-tracked, writes LDS directly)
// ---------------------------------------------------------------------------
__device__ __forceinline__ void async_copy_b128(unsigned lds_off, const void* gaddr) {
  asm volatile("global_load_async_to_lds_b128 %0, %1, off"
               :: "v"(lds_off), "v"(gaddr)
               : "memory");
}
__device__ __forceinline__ void wait_async0() {
  asm volatile("s_wait_asynccnt 0x0" ::: "memory");
}
__device__ __forceinline__ unsigned lds_off32(const void* p) {
  // generic LDS pointer: aperture in [63:32], LDS byte offset in [31:0]
  return (unsigned)(unsigned long long)p;
}

// ---------------------------------------------------------------------------
// fp32 -> bf16 elementwise conversion
// ---------------------------------------------------------------------------
__global__ void f32_to_bf16_k(const float* __restrict__ src, __bf16* __restrict__ dst, int n) {
  int i = blockIdx.x * blockDim.x + threadIdx.x;
  int stride = gridDim.x * blockDim.x;
  for (; i < n; i += stride) dst[i] = (__bf16)src[i];
}

// ---------------------------------------------------------------------------
// spectral norm pieces (deterministic: no float atomics)
// ---------------------------------------------------------------------------
// t[j] = sum_i W[i,j] * u[i]
__global__ void sn_wt_u_k(const float* __restrict__ W, const float* __restrict__ u,
                          float* __restrict__ t, int rows, int cols) {
  int j = blockIdx.x * blockDim.x + threadIdx.x;
  if (j >= cols) return;
  float s = 0.f;
  for (int i = 0; i < rows; ++i) s += W[(size_t)i * cols + j] * u[i];
  t[j] = s;
}

// y[i] = sum_j W[i,j] * (t[j]/(sqrt(nt2)+eps))
__global__ void sn_w_v_k(const float* __restrict__ W, const float* __restrict__ t,
                         const float* __restrict__ nt2, float* __restrict__ y,
                         int rows, int cols) {
  int i = blockIdx.x * blockDim.x + threadIdx.x;
  if (i >= rows) return;
  float inv = 1.0f / (sqrtf(*nt2) + EPS_C);
  float s = 0.f;
  const float* __restrict__ Wr = W + (size_t)i * cols;
  for (int j = 0; j < cols; ++j) s += Wr[j] * (t[j] * inv);
  y[i] = s;
}

// out = sum v[i]^2  (single workgroup, fixed-order tree reduction: deterministic)
__global__ void reduce_sq_k(const float* __restrict__ v, int n, float* __restrict__ out) {
  __shared__ float s[256];
  float acc = 0.f;
  for (int i = threadIdx.x; i < n; i += 256) acc += v[i] * v[i];
  s[threadIdx.x] = acc;
  __syncthreads();
  for (int w = 128; w > 0; w >>= 1) {
    if (threadIdx.x < w) s[threadIdx.x] += s[threadIdx.x + w];
    __syncthreads();
  }
  if (threadIdx.x == 0) *out = s[0];
}

// sigma = ny2/(sqrt(ny2)+eps) ; inv_sigma = 1/sigma
__global__ void sn_finish_k(const float* __restrict__ ny2, float* __restrict__ inv_sigma) {
  float n2 = *ny2;
  float sigma = n2 / (sqrtf(n2) + EPS_C);
  *inv_sigma = 1.0f / sigma;
}

// ---------------------------------------------------------------------------
// WMMA GEMM: C[m,n] = sum_k A[m,k] * Bw[n,k]   (A: MxK bf16, Bw: NxK bf16)
// Block: 256 threads (8 wave32s), tile 64(M) x 128(N), K chunk 32,
// double-buffered LDS filled by async global->LDS copies.
// Waves: wm = wid&1 (2 M-frags of 16), wn = wid>>1 (2 N-frags of 16).
// Each wave: 2 A-frags x 2 B-frags -> 4 WMMAs from 8 ds_load_b128.
// ---------------------------------------------------------------------------
__launch_bounds__(256)
__global__ void wmma_gemm_k(const __bf16* __restrict__ A,
                            const __bf16* __restrict__ Bw,
                            int M, int N, int K, int mode,
                            const float* __restrict__ bias,
                            const float* __restrict__ addend,
                            const float* __restrict__ h_prev,
                            const float* __restrict__ scale_p,
                            float* __restrict__ Cf,
                            __bf16* __restrict__ Cbf) {
  constexpr int BM = 64, BN = 128, BK = 32;
  constexpr int LDT = BK + 8;  // padded LDS row stride (bf16) to spread banks

  __shared__ alignas(16) __bf16 lA[2][BM * LDT];
  __shared__ alignas(16) __bf16 lB[2][BN * LDT];

  const int tid  = threadIdx.x;
  const int wid  = tid >> 5;
  const int lane = tid & 31;
  const int wm   = wid & 1;     // 0..1 : rows wm*32 .. wm*32+31 (2 m-frags)
  const int wn   = wid >> 1;    // 0..3 : cols wn*32 .. wn*32+31 (2 n-frags)
  const int m0   = blockIdx.y * BM;
  const int n0   = blockIdx.x * BN;

  const int hi  = lane >> 4;    // 0 -> lanes 0-15, 1 -> lanes 16-31
  const int l15 = lane & 15;

  // staging assignments: per thread 1x16B for A, 2x16B for B per K chunk
  const int arow = tid >> 2;          // 0..63
  const int akg  = (tid & 3) * 8;     // 0,8,16,24 (elems)
  const int brow = tid >> 1;          // 0..127
  const int bkg  = (tid & 1) * 16;    // 0,16 (elems)
  int gnb = n0 + brow; if (gnb >= N) gnb = N - 1;  // clamp: masked at store

  const __bf16* gA = A  + (size_t)(m0 + arow) * K + akg;
  const __bf16* gB = Bw + (size_t)gnb * K + bkg;

  auto stage = [&](int kc, int buf) {
    async_copy_b128(lds_off32(&lA[buf][arow * LDT + akg]), gA + kc);
    async_copy_b128(lds_off32(&lB[buf][brow * LDT + bkg]), gB + kc);
    async_copy_b128(lds_off32(&lB[buf][brow * LDT + bkg + 8]), gB + kc + 8);
  };

  v8f acc[2][2] = {};

  const int nk = K / BK;
  stage(0, 0);  // prologue
  for (int kt = 0; kt < nk; ++kt) {
    const int cur = kt & 1;
    wait_async0();       // my async writes into buf 'cur' are complete
    __syncthreads();     // everyone's writes visible; everyone done reading buf 'cur^1'
    if (kt + 1 < nk) stage((kt + 1) * BK, cur ^ 1);  // overlap with compute

    // ---- A fragments (16x32 bf16): lanes 0-15 K0..7 & 16..23 of row M=lane;
    //      lanes 16-31 K8..15 & 24..31 of row M=lane-16 ----
    v16bf af[2];
#pragma unroll
    for (int t = 0; t < 2; ++t) {
      const __bf16* pa = &lA[cur][(wm * 32 + t * 16 + l15) * LDT + hi * 8];
      v8bf a0 = *(const v8bf*)pa;
      v8bf a1 = *(const v8bf*)(pa + 16);
      af[t] = __builtin_shufflevector(a0, a1,
          0,1,2,3,4,5,6,7,8,9,10,11,12,13,14,15);
    }
    // ---- B fragments (32x16 bf16): lane holds column n=l15;
    //      lanes 0-15 K0..7 & 16..23, lanes 16-31 K8..15 & 24..31 ----
    v16bf bfr[2];
#pragma unroll
    for (int j = 0; j < 2; ++j) {
      const __bf16* pb = &lB[cur][(wn * 32 + j * 16 + l15) * LDT + hi * 8];
      v8bf b0 = *(const v8bf*)pb;
      v8bf b1 = *(const v8bf*)(pb + 16);
      bfr[j] = __builtin_shufflevector(b0, b1,
          0,1,2,3,4,5,6,7,8,9,10,11,12,13,14,15);
    }
#pragma unroll
    for (int t = 0; t < 2; ++t)
#pragma unroll
      for (int j = 0; j < 2; ++j)
        acc[t][j] = __builtin_amdgcn_wmma_f32_16x16x32_bf16(
            false, af[t], false, bfr[j], (short)0, acc[t][j], false, false);
  }

  // ---- epilogue ----
  // C/D layout: lanes 0-15 -> N=lane, M=r; lanes 16-31 -> N=lane-16, M=8+r
  float scale = (mode == EPI_XPROJ) ? *scale_p : 1.0f;
#pragma unroll
  for (int t = 0; t < 2; ++t) {
#pragma unroll
    for (int j = 0; j < 2; ++j) {
      int gn = n0 + wn * 32 + j * 16 + l15;
      if (gn >= N) continue;
      float bv = bias[gn];
#pragma unroll
      for (int r = 0; r < 8; ++r) {
        int gm = m0 + wm * 32 + t * 16 + hi * 8 + r;
        size_t idx = (size_t)gm * N + gn;
        float v = acc[t][j][r];
        if (mode == EPI_XPROJ) {
          Cf[idx] = v * scale + bv;
        } else if (mode == EPI_STEP) {
          float hn = (1.0f - ALPHA_C) * h_prev[idx]
                   + ALPHA_C * tanhf(addend[idx] + v + bv);
          Cf[idx] = hn;
          Cbf[idx] = (__bf16)hn;
        } else {
          Cf[idx] = v + bv;
        }
      }
    }
  }
}

// ---------------------------------------------------------------------------
// host-side launch
// ---------------------------------------------------------------------------
extern "C" void kernel_launch(void* const* d_in, const int* in_sizes, int n_in,
                              void* d_out, int out_size, void* d_ws, size_t ws_size,
                              hipStream_t stream) {
  (void)in_sizes; (void)n_in; (void)out_size; (void)ws_size;

  const int B = 256, D_IN = 1024, D_H = 2048, D_OUT = 1000, STEPS = 30;

  const float* x      = (const float*)d_in[0];
  const float* W_in_w = (const float*)d_in[1];
  const float* W_in_b = (const float*)d_in[2];
  const float* u      = (const float*)d_in[3];
  const float* W0     = (const float*)d_in[4];
  const float* b0     = (const float*)d_in[5];
  const float* W1     = (const float*)d_in[6];
  const float* b1     = (const float*)d_in[7];
  const float* W2     = (const float*)d_in[8];
  const float* b2     = (const float*)d_in[9];
  const float* head_w = (const float*)d_in[10];
  const float* head_b = (const float*)d_in[11];
  float* out = (float*)d_out;

  // ---- workspace bump allocator (256B aligned) ----
  char* ws = (char*)d_ws;
  size_t off = 0;
  auto alloc = [&](size_t bytes) -> char* {
    char* p = ws + off;
    off += (bytes + 255) & ~(size_t)255;
    return p;
  };

  float*  x_proj  = (float*)alloc((size_t)B * D_H * 4);
  float*  hA      = (float*)alloc((size_t)B * D_H * 4);
  float*  hB      = (float*)alloc((size_t)B * D_H * 4);
  __bf16* hA_bf   = (__bf16*)alloc((size_t)B * D_H * 2);
  __bf16* hB_bf   = (__bf16*)alloc((size_t)B * D_H * 2);
  __bf16* Win_bf  = (__bf16*)alloc((size_t)D_H * D_IN * 2);
  __bf16* x_bf    = (__bf16*)alloc((size_t)B * D_IN * 2);
  __bf16* W0_bf   = (__bf16*)alloc((size_t)D_H * D_H * 2);
  __bf16* W1_bf   = (__bf16*)alloc((size_t)D_H * D_H * 2);
  __bf16* W2_bf   = (__bf16*)alloc((size_t)D_H * D_H * 2);
  __bf16* hw_bf   = (__bf16*)alloc((size_t)D_OUT * D_H * 2);
  float*  t_vec   = (float*)alloc((size_t)D_IN * 4);
  float*  y_vec   = (float*)alloc((size_t)D_H * 4);
  float*  nt2     = (float*)alloc(4);
  float*  ny2     = (float*)alloc(4);
  float*  inv_sig = (float*)alloc(4);

  // ---- zero-init state h0 = 0 ----
  hipMemsetAsync(hA, 0, (size_t)B * D_H * 4, stream);
  hipMemsetAsync(hA_bf, 0, (size_t)B * D_H * 2, stream);

  // ---- fp32 -> bf16 conversions ----
  auto cvt = [&](const float* s, __bf16* d, int n) {
    int blocks = (n + 255) / 256;
    if (blocks > 1024) blocks = 1024;
    f32_to_bf16_k<<<blocks, 256, 0, stream>>>(s, d, n);
  };
  cvt(W_in_w, Win_bf, D_H * D_IN);
  cvt(x, x_bf, B * D_IN);
  cvt(W0, W0_bf, D_H * D_H);
  cvt(W1, W1_bf, D_H * D_H);
  cvt(W2, W2_bf, D_H * D_H);
  cvt(head_w, hw_bf, D_OUT * D_H);

  // ---- spectral norm: inv_sigma = 1 / || W normalize(W^T u) || ----
  sn_wt_u_k<<<(D_IN + 255) / 256, 256, 0, stream>>>(W_in_w, u, t_vec, D_H, D_IN);
  reduce_sq_k<<<1, 256, 0, stream>>>(t_vec, D_IN, nt2);
  sn_w_v_k<<<(D_H + 255) / 256, 256, 0, stream>>>(W_in_w, t_vec, nt2, y_vec, D_H, D_IN);
  reduce_sq_k<<<1, 256, 0, stream>>>(y_vec, D_H, ny2);
  sn_finish_k<<<1, 1, 0, stream>>>(ny2, inv_sig);

  // ---- x_proj = (x @ W_in^T) * inv_sigma + W_in_b ----
  {
    dim3 grid(D_H / 128, B / 64);
    wmma_gemm_k<<<grid, 256, 0, stream>>>(x_bf, Win_bf, B, D_H, D_IN, EPI_XPROJ,
                                          W_in_b, nullptr, nullptr, inv_sig,
                                          x_proj, nullptr);
  }

  // ---- recurrent steps: h = (1-a)h + a*tanh(x_proj + h @ W^T + b) ----
  const __bf16* Wl_bf[3] = {W0_bf, W1_bf, W2_bf};
  const float*  bl[3]    = {b0, b1, b2};
  float*  hc = hA;  __bf16* hc_bf = hA_bf;
  float*  hn = hB;  __bf16* hn_bf = hB_bf;
  dim3 grid_step(D_H / 128, B / 64);
  for (int s = 0; s < STEPS; ++s) {
    for (int l = 0; l < 3; ++l) {
      wmma_gemm_k<<<grid_step, 256, 0, stream>>>(hc_bf, Wl_bf[l], B, D_H, D_H,
                                                 EPI_STEP, bl[l], x_proj, hc,
                                                 nullptr, hn, hn_bf);
      float* tf = hc; hc = hn; hn = tf;
      __bf16* tb = hc_bf; hc_bf = hn_bf; hn_bf = tb;
    }
  }

  // ---- head: out = h @ head_w^T + head_b ----
  {
    dim3 grid((D_OUT + 127) / 128, B / 64);
    wmma_gemm_k<<<grid, 256, 0, stream>>>(hc_bf, hw_bf, B, D_OUT, D_H, EPI_HEAD,
                                          head_b, nullptr, nullptr, nullptr,
                                          out, nullptr);
  }
}
